// RNN_28398323761224
// MI455X (gfx1250) — compile-verified, matrix operand order checked by reading, not run on hardware
//
#include <hip/hip_runtime.h>

// ---------------- problem constants ----------------
#define VOCAB 500
#define EMBED 80
#define HID   512
#define CHN   128
#define BB    128
#define LL    784
#define TT    256
#define G4    2048          // 4*HID
#define WIHC  592           // HID + EMBED
#define HSTR  520           // padded LDS row stride in bf16 elems (bank spread)

// ---------------- types ----------------
typedef __attribute__((ext_vector_type(16))) __bf16 v16bf;
typedef __attribute__((ext_vector_type(8)))  float  v8f;

struct alignas(16) F4  { float x, y, z, w; };
struct F42 { F4 lo, hi; };

static_assert(sizeof(F42) == sizeof(v16bf), "bitcast size");

__device__ inline unsigned short f2bf(float x) {
    unsigned u = __float_as_uint(x);
    u += 0x7FFFu + ((u >> 16) & 1u);      // round to nearest even
    return (unsigned short)(u >> 16);
}
__device__ inline float sigf(float x) { return 1.0f / (1.0f + __expf(-x)); }
__device__ inline float tanh_fast(float x) {
    float ax = fabsf(x);
    float e  = __expf(-2.0f * ax);
    float r  = (1.0f - e) / (1.0f + e);
    return copysignf(r, x);
}

// ---------------- precompute kernels ----------------

// mean over L: latent [B, L, C] -> mean [B, C]
__global__ __launch_bounds__(CHN) void mean_kernel(const float* __restrict__ lat,
                                                   float* __restrict__ mean) {
    int b = blockIdx.x, j = threadIdx.x;
    const float* p = lat + (size_t)b * LL * CHN + j;
    float s = 0.f;
    for (int l = 0; l < LL; ++l) s += p[(size_t)l * CHN];
    mean[b * CHN + j] = s * (1.0f / (float)LL);
}

// h0/c0/o0 = tanh(mean @ W.T + b)
__global__ __launch_bounds__(HID) void init_kernel(const float* __restrict__ mean,
                                                   const float* __restrict__ Wh, const float* __restrict__ bh,
                                                   const float* __restrict__ Wc, const float* __restrict__ bc,
                                                   const float* __restrict__ Wo, const float* __restrict__ bo,
                                                   float* __restrict__ h0, float* __restrict__ c0,
                                                   float* __restrict__ o0) {
    __shared__ float sm[CHN];
    int b = blockIdx.x, j = threadIdx.x;
    if (j < CHN) sm[j] = mean[b * CHN + j];
    __syncthreads();
    float sh = bh[j], sc = bc[j], so = bo[j];
    const float* wh = Wh + (size_t)j * CHN;
    const float* wc = Wc + (size_t)j * CHN;
    const float* wo = Wo + (size_t)j * CHN;
    for (int k = 0; k < CHN; ++k) {
        float m = sm[k];
        sh += m * wh[k]; sc += m * wc[k]; so += m * wo[k];
    }
    h0[b * HID + j] = tanh_fast(sh);
    c0[b * HID + j] = tanh_fast(sc);
    o0[b * HID + j] = tanh_fast(so);
}

// gates_base[b,g] = b_ih[g] + b_hh[g] + dot(o0[b,:], W_ih[g, 0:512])
__global__ __launch_bounds__(256) void gbase_kernel(const float* __restrict__ o0,
                                                    const float* __restrict__ W_ih,
                                                    const float* __restrict__ b_ih,
                                                    const float* __restrict__ b_hh,
                                                    float* __restrict__ gbase) {
    __shared__ float so[HID];
    int b = blockIdx.x, tid = threadIdx.x;
    int g = blockIdx.y * 256 + tid;
    so[tid]       = o0[b * HID + tid];
    so[tid + 256] = o0[b * HID + tid + 256];
    __syncthreads();
    const float* w = W_ih + (size_t)g * WIHC;
    float s = b_ih[g] + b_hh[g];
    for (int k = 0; k < HID; ++k) s += so[k] * w[k];
    gbase[(size_t)b * G4 + g] = s;
}

// emb_proj[v,g] = dot(embedding[v,:], W_ih[g, 512:592])
__global__ __launch_bounds__(256) void eproj_kernel(const float* __restrict__ emb,
                                                    const float* __restrict__ W_ih,
                                                    float* __restrict__ ep) {
    __shared__ float se[EMBED];
    int v = blockIdx.x, tid = threadIdx.x;
    int g = blockIdx.y * 256 + tid;
    if (tid < EMBED) se[tid] = emb[v * EMBED + tid];
    __syncthreads();
    const float* w = W_ih + (size_t)g * WIHC + HID;
    float s = 0.f;
    for (int e = 0; e < EMBED; ++e) s += se[e] * w[e];
    ep[(size_t)v * G4 + g] = s;
}

// Pack W_hh (fp32, row-major [2048,512]) into bf16 WMMA-B-tile order:
// chunk (nt, kk, lane, half) -> 8 consecutive bf16.
// Packed byte offset = nt*16384 + kk*1024 + lane*32 + half*16, so inside the
// LSTM kernel all tile selection folds into the load instruction's 24-bit
// immediate offset on a single per-thread base pointer.
__global__ __launch_bounds__(256) void pack_whh_kernel(const float* __restrict__ w,
                                                       unsigned short* __restrict__ o) {
    int tid  = blockIdx.x * 256 + threadIdx.x;   // 0 .. 131071
    int half = tid & 1;
    int l    = (tid >> 1) & 31;
    int kk   = (tid >> 6) & 15;
    int nt   = tid >> 10;                        // 0..127
    int row  = nt * 16 + (l & 15);               // W_hh row = gate column
    int col  = kk * 32 + ((l >= 16) ? 8 : 0) + half * 16;
    const float* src = w + (size_t)row * HID + col;
    unsigned short* dst = o + ((size_t)((nt * 16 + kk) * 32 + l)) * 16 + half * 8;
    for (int i = 0; i < 8; ++i) dst[i] = f2bf(src[i]);
}

// ---------------- persistent LSTM kernel ----------------
// 8 blocks (16 batch rows each) x 512 threads (16 wave32).
// Wave w owns hidden cols [w*32, w*32+32): 2 N-tiles per gate, 8 acc tiles.
__global__ __launch_bounds__(512) void lstm_kernel(const unsigned short* __restrict__ whh,
                                                   const float* __restrict__ gbase,
                                                   const float* __restrict__ eproj,
                                                   const int* __restrict__ formulas,
                                                   const float* __restrict__ h0,
                                                   const float* __restrict__ c0,
                                                   float* __restrict__ out) {
    __shared__ unsigned short hbuf[2][16 * HSTR];
    __shared__ int tokb[2][16];

    const int tid   = threadIdx.x;
    const int w     = tid >> 5;
    const int lane  = tid & 31;
    const bool hi   = lane >= 16;
    const int lc    = hi ? lane - 16 : lane;     // tile-local row/col index
    const int bbase = blockIdx.x * 16;
    const int colbase = w * 32;

    // single per-thread B base pointer; tiles are immediate offsets off it
    const char* bb = (const char*)whh + (size_t)w * 32768 + (size_t)lane * 32;

    // stage h0 into hbuf[0] as bf16
    for (int i = tid; i < 16 * HID; i += blockDim.x) {
        int r = i >> 9, c = i & (HID - 1);
        hbuf[0][r * HSTR + c] = f2bf(h0[(size_t)(bbase + r) * HID + c]);
    }

    // c state in registers, laid out exactly like the WMMA D tiles
    v8f cst[2];
    for (int p = 0; p < 2; ++p)
        for (int r = 0; r < 8; ++r) {
            int rl = hi ? r + 8 : r;
            cst[p][r] = c0[(size_t)(bbase + rl) * HID + colbase + p * 16 + lc];
        }

    const int kofs = hi ? 8 : 0;   // K-offset of this half-wave (A striping)

    for (int t = 0; t < TT; ++t) {
        const int cur = t & 1, nxt = cur ^ 1;
        if (tid < 16) tokb[cur][tid] = formulas[(size_t)(bbase + tid) * TT + t];
        __syncthreads();
        // Full memory clobber: forbid LICM from hoisting the (t-invariant)
        // W_hh / gates_base loads out of the time loop (round-1 asm showed
        // the whole B working set hoisted and spilled to scratch).
        asm volatile("" ::: "memory");

        // warm emb_proj rows for this step's tokens (global_prefetch_b8)
        {
            int tk0 = tokb[cur][hi ? 8 : 0];
            const float* ep0 = eproj + (size_t)tk0 * G4 + colbase + lc;
            __builtin_prefetch(ep0,        0, 0);
            __builtin_prefetch(ep0 + 512,  0, 0);
            __builtin_prefetch(ep0 + 1024, 0, 0);
            __builtin_prefetch(ep0 + 1536, 0, 0);
        }

        v8f acc[4][2];
        for (int g = 0; g < 4; ++g)
            for (int p = 0; p < 2; ++p)
                for (int r = 0; r < 8; ++r) acc[g][p][r] = 0.f;

        const unsigned short* hrow = &hbuf[cur][0];
        const int aoff = lc * HSTR + kofs;   // A: lane row = lc

#pragma unroll 4
        for (int kk = 0; kk < 16; ++kk) {
            // A tile (shared by all 8 wmma of this k-step): two ds_load_b128
            const F4* ap = reinterpret_cast<const F4*>(hrow + aoff + kk * 32);
            F42 araw; araw.lo = ap[0]; araw.hi = ap[2];   // K +0..7, +16..23 (or +8.., +24..)
            v16bf av = __builtin_bit_cast(v16bf, araw);

            for (int g = 0; g < 4; ++g) {
                for (int p = 0; p < 2; ++p) {
                    // packed B tile: base + immediate offset, 2 coalesced b128
                    const char* bptr = bb + ((size_t)g * 524288 +
                                             (size_t)p * 16384 +
                                             (size_t)kk * 1024);
                    F42 braw;
                    braw.lo = *reinterpret_cast<const F4*>(bptr);
                    braw.hi = *reinterpret_cast<const F4*>(bptr + 16);
                    v16bf bv = __builtin_bit_cast(v16bf, braw);
                    acc[g][p] = __builtin_amdgcn_wmma_f32_16x16x32_bf16(
                        false, av, false, bv, (short)0, acc[g][p], false, false);
                }
            }
        }

        // elementwise LSTM on the wave's 32 hidden units x 16 batch rows
        for (int p = 0; p < 2; ++p) {
            const int col = colbase + p * 16 + lc;
            for (int r = 0; r < 8; ++r) {
                const int rl = hi ? r + 8 : r;
                const int b  = bbase + rl;
                const int tk = tokb[cur][rl];
                const float* gb = gbase + (size_t)b  * G4;
                const float* ep = eproj + (size_t)tk * G4;

                float iv = acc[0][p][r] + gb[col]        + ep[col];
                float fv = acc[1][p][r] + gb[512  + col] + ep[512  + col];
                float gv = acc[2][p][r] + gb[1024 + col] + ep[1024 + col];
                float ov = acc[3][p][r] + gb[1536 + col] + ep[1536 + col];

                float c = sigf(fv) * cst[p][r] + sigf(iv) * tanh_fast(gv);
                float h = sigf(ov) * tanh_fast(c);
                cst[p][r] = c;

                hbuf[nxt][rl * HSTR + col] = f2bf(h);
                out[((size_t)b * TT + t) * HID + col] = h;
            }
        }
        // next iteration's __syncthreads protects hbuf/tok reuse
    }
}

// ---------------- launcher ----------------
extern "C" void kernel_launch(void* const* d_in, const int* in_sizes, int n_in,
                              void* d_out, int out_size, void* d_ws, size_t ws_size,
                              hipStream_t stream) {
    (void)in_sizes; (void)n_in; (void)out_size; (void)ws_size;

    const float* latent    = (const float*)d_in[0];
    const int*   formulas  = (const int*)  d_in[1];
    const float* embedding = (const float*)d_in[2];
    const float* W_ih      = (const float*)d_in[3];
    const float* b_ih      = (const float*)d_in[4];
    const float* W_hh      = (const float*)d_in[5];
    const float* b_hh      = (const float*)d_in[6];
    const float* Wh        = (const float*)d_in[7];
    const float* bh        = (const float*)d_in[8];
    const float* Wc        = (const float*)d_in[9];
    const float* bc        = (const float*)d_in[10];
    const float* Wo        = (const float*)d_in[11];
    const float* bo        = (const float*)d_in[12];
    float* out = (float*)d_out;

    char* ws = (char*)d_ws;
    const size_t OFF_WHH   = 0;                                   // 2048*512*2  = 2,097,152
    const size_t OFF_EPROJ = OFF_WHH   + (size_t)G4 * HID * 2;    // 500*2048*4  = 4,096,000
    const size_t OFF_GBASE = OFF_EPROJ + (size_t)VOCAB * G4 * 4;  // 128*2048*4  = 1,048,576
    const size_t OFF_MEAN  = OFF_GBASE + (size_t)BB * G4 * 4;     // 128*128*4
    const size_t OFF_H0    = OFF_MEAN  + (size_t)BB * CHN * 4;    // 128*512*4
    const size_t OFF_C0    = OFF_H0    + (size_t)BB * HID * 4;
    const size_t OFF_O0    = OFF_C0    + (size_t)BB * HID * 4;

    unsigned short* whh_bf = (unsigned short*)(ws + OFF_WHH);
    float* eproj = (float*)(ws + OFF_EPROJ);
    float* gbase = (float*)(ws + OFF_GBASE);
    float* meanb = (float*)(ws + OFF_MEAN);
    float* h0    = (float*)(ws + OFF_H0);
    float* c0    = (float*)(ws + OFF_C0);
    float* o0    = (float*)(ws + OFF_O0);

    mean_kernel <<<BB, CHN, 0, stream>>>(latent, meanb);
    init_kernel <<<BB, HID, 0, stream>>>(meanb, Wh, bh, Wc, bc, Wo, bo, h0, c0, o0);
    gbase_kernel<<<dim3(BB, G4 / 256), 256, 0, stream>>>(o0, W_ih, b_ih, b_hh, gbase);
    eproj_kernel<<<dim3(VOCAB, G4 / 256), 256, 0, stream>>>(embedding, W_ih, eproj);
    pack_whh_kernel<<<(G4 * HID / 8) / 256, 256, 0, stream>>>(W_hh, whh_bf);
    lstm_kernel <<<BB / 16, 512, 0, stream>>>(whh_bf, gbase, eproj, formulas, h0, c0, out);
}
